// Bert4KGModel_65051574665427
// MI455X (gfx1250) — compile-verified, matrix operand order checked by reading, not run on hardware
//
#include <hip/hip_runtime.h>
#include <hip/hip_bf16.h>

// ---------------------------------------------------------------------------
// Types for CDNA5 WMMA
// ---------------------------------------------------------------------------
typedef _Float16 h16;
typedef __attribute__((ext_vector_type(16))) _Float16 v16h;
typedef __attribute__((ext_vector_type(8)))  _Float16 v8h;
typedef __attribute__((ext_vector_type(2)))  _Float16 v2h;
typedef __attribute__((ext_vector_type(8)))  float    v8f;
typedef int v4i __attribute__((vector_size(16)));

#define GLOBAL_AS __attribute__((address_space(1)))
#define LDS_AS    __attribute__((address_space(3)))

#define BM 128
#define BN 64
#define BK 32

// Async global->LDS path (CDNA5 GLOBAL_LOAD_ASYNC_TO_LDS_B128, ASYNCcnt).
// Guarded: falls back to load->VGPR->ds_store if the toolchain lacks the builtin.
#if defined(__gfx1250__) && \
    __has_builtin(__builtin_amdgcn_global_load_async_to_lds_b128) && \
    __has_builtin(__builtin_amdgcn_s_wait_asynccnt)
#define ASYNC_LDS 1
#else
#define ASYNC_LDS 0
#endif

// Load a 16x32 f16 WMMA fragment row for this lane from an LDS tile row.
// ISA 7.12.2 16-bit A-matrix layout:
//   lanes 0-15 : VGPR v(0..3) -> K = 2v,2v+1 ; v(4..7) -> K = 2v+8, 2v+9
//   lanes 16-31: same + 8
__device__ __forceinline__ v16h load_frag_row(const h16* rowp, int lhalf) {
  v16h f;
#pragma unroll
  for (int v = 0; v < 8; ++v) {
    int kb = ((v < 4) ? (2 * v) : (2 * v + 8)) + lhalf * 8;
    v2h p = *(const v2h*)(rowp + kb);
    f[2 * v]     = p[0];
    f[2 * v + 1] = p[1];
  }
  return f;
}

// ---------------------------------------------------------------------------
// Generic batched GEMM:  C[z][M,N] = alpha * (A[z][M,K] * B[z][N,K]^eff + bias)
// A is K-contiguous (lda = row stride). B element (n,k) at n*ldbn + k*ldbk.
// z -> (b = z / Hn, h = z % Hn) with separate b/h strides for A, B, C.
// Requires M%128==0, N%64==0, K%32==0 (true for every call in this model).
// ---------------------------------------------------------------------------
__global__ __launch_bounds__(256) void gemm_wmma(
    const h16* __restrict__ A, const h16* __restrict__ Bm,
    const float* __restrict__ bias,
    float* __restrict__ C, h16* __restrict__ C16,
    int M, int N, int K,
    int lda, int ldbn, int ldbk, int ldc,
    long sAb, long sAh, long sBb, long sBh, long sCb, long sCh,
    int Hn, float alpha, int relu)
{
  __shared__ h16 As[BM][BK + 8];
  __shared__ h16 Bs[BN][BK + 8];

  const int z  = blockIdx.z;
  const int b  = z / Hn;
  const int hh = z % Hn;
  const h16* Ab = A  + (long)b * sAb + (long)hh * sAh;
  const h16* Bb = Bm + (long)b * sBb + (long)hh * sBh;
  const long coff = (long)b * sCb + (long)hh * sCh;

  const int bm = blockIdx.y * BM;
  const int bn = blockIdx.x * BN;
  const int tid  = threadIdx.x;
  const int wave = tid >> 5;
  const int lane = tid & 31;
  const int wm = wave >> 1;      // 0..3  (M direction)
  const int wn = wave & 1;       // 0..1  (N direction)
  const int lhalf = lane >> 4;   // 0/1
  const int lmod  = lane & 15;

  const v8f vzero = {0.f, 0.f, 0.f, 0.f, 0.f, 0.f, 0.f, 0.f};
  v8f acc[2][2];
#pragma unroll
  for (int mi = 0; mi < 2; ++mi)
#pragma unroll
    for (int ni = 0; ni < 2; ++ni) acc[mi][ni] = vzero;

  // A tile: 128x32 halves, 256 threads -> 16 halves each (2 x 16B chunks)
  const int arow = tid >> 1;
  const int acol = (tid & 1) * 16;
  // B tile: 64x32 halves, 256 threads -> 8 halves each (1 x 16B chunk)
  const int brow = tid >> 2;
  const int bcol = (tid & 3) * 8;

  for (int k0 = 0; k0 < K; k0 += BK) {
    const h16* asrc = Ab + (long)(bm + arow) * lda + k0 + acol;
    __builtin_prefetch(asrc + BK, 0, 0);   // global_prefetch_b8: next K tile
#if ASYNC_LDS
    // One immediate offset applies to BOTH the LDS and global address (ISA 8 4.4)
    __builtin_amdgcn_global_load_async_to_lds_b128(
        (GLOBAL_AS v4i*)asrc, (LDS_AS v4i*)&As[arow][acol], 0, 0);
    __builtin_amdgcn_global_load_async_to_lds_b128(
        (GLOBAL_AS v4i*)asrc, (LDS_AS v4i*)&As[arow][acol], 16, 0);
#else
    *(v8h*)&As[arow][acol]     = *(const v8h*)asrc;
    *(v8h*)&As[arow][acol + 8] = *(const v8h*)(asrc + 8);
#endif
    if (ldbk == 1) {
      const h16* bsrc = Bb + (long)(bn + brow) * ldbn + k0 + bcol;
#if ASYNC_LDS
      __builtin_amdgcn_global_load_async_to_lds_b128(
          (GLOBAL_AS v4i*)bsrc, (LDS_AS v4i*)&Bs[brow][bcol], 0, 0);
#else
      *(v8h*)&Bs[brow][bcol] = *(const v8h*)bsrc;
#endif
    } else {
      const h16* bsrc = Bb + (long)(bn + brow) * ldbn + (long)(k0 + bcol) * ldbk;
#pragma unroll
      for (int i = 0; i < 8; ++i) Bs[brow][bcol + i] = bsrc[(long)i * ldbk];
    }
#if ASYNC_LDS
    __builtin_amdgcn_s_wait_asynccnt(0);
#endif
    __syncthreads();

    v16h af[2], bf[2];
#pragma unroll
    for (int mi = 0; mi < 2; ++mi)
      af[mi] = load_frag_row(&As[wm * 32 + mi * 16 + lmod][0], lhalf);
#pragma unroll
    for (int ni = 0; ni < 2; ++ni)
      bf[ni] = load_frag_row(&Bs[wn * 32 + ni * 16 + lmod][0], lhalf);

#pragma unroll
    for (int mi = 0; mi < 2; ++mi)
#pragma unroll
      for (int ni = 0; ni < 2; ++ni)
        acc[mi][ni] = __builtin_amdgcn_wmma_f32_16x16x32_f16(
            false, af[mi], false, bf[ni], (short)0, acc[mi][ni], false, false);

    __syncthreads();
  }

  float* Cp  = C   ? (C + coff)   : nullptr;
  h16*  C16p = C16 ? (C16 + coff) : nullptr;
#pragma unroll
  for (int mi = 0; mi < 2; ++mi) {
#pragma unroll
    for (int ni = 0; ni < 2; ++ni) {
      const int col  = bn + wn * 32 + ni * 16 + lmod;
      const int row0 = bm + wm * 32 + mi * 16 + lhalf * 8;
      const float bv = bias ? bias[col] : 0.0f;
#pragma unroll
      for (int r = 0; r < 8; ++r) {
        float v = (acc[mi][ni][r] + bv) * alpha;
        if (relu) v = fmaxf(v, 0.0f);
        const long idx = (long)(row0 + r) * ldc + col;
        if (Cp)   Cp[idx]   = v;
        if (C16p) C16p[idx] = (h16)v;
      }
    }
  }
}

// ---------------------------------------------------------------------------
// fp32 -> f16 conversion (grid-stride)
// ---------------------------------------------------------------------------
__global__ void cvt16(const float* __restrict__ in, h16* __restrict__ out, long n) {
  long i  = (long)blockIdx.x * blockDim.x + threadIdx.x;
  long st = (long)gridDim.x * blockDim.x;
  for (; i < n; i += st) out[i] = (h16)in[i];
}

// ---------------------------------------------------------------------------
// f16 transpose  v[b,k,d] -> vt[b,d,k]  (32x32 LDS tiles, both sides coalesced)
// ---------------------------------------------------------------------------
__global__ __launch_bounds__(256) void transpose_kd(
    const h16* __restrict__ v, h16* __restrict__ vt, int Kl, int D)
{
  __shared__ h16 t[32][33];
  const int b  = blockIdx.z;
  const int k0 = blockIdx.x * 32;
  const int d0 = blockIdx.y * 32;
  const int tx = threadIdx.x & 31;
  const int ty = threadIdx.x >> 5;   // 0..7
  const h16* vb  = v  + (long)b * Kl * D;
  h16*       vtb = vt + (long)b * D * Kl;
#pragma unroll
  for (int i = 0; i < 32; i += 8)
    t[ty + i][tx] = vb[(long)(k0 + ty + i) * D + d0 + tx];
  __syncthreads();
#pragma unroll
  for (int i = 0; i < 32; i += 8)
    vtb[(long)(d0 + ty + i) * Kl + k0 + tx] = t[tx][ty + i];
}

// ---------------------------------------------------------------------------
// Masked softmax over last dim of S[B,H,Q,Kl]; emits f16 probabilities.
// mask3d: mask[b,q,k] (causal) else mask[b,k] (cross).  NEG_INF = -1e20.
// ---------------------------------------------------------------------------
__global__ __launch_bounds__(256) void masked_softmax(
    const float* __restrict__ S, const int* __restrict__ mask,
    h16* __restrict__ P, int Kl, int Q, int mask3d)
{
  const int q  = blockIdx.x;
  const int hh = blockIdx.y;
  const int b  = blockIdx.z;
  const int Hn = gridDim.y;
  const long rowoff = (((long)b * Hn + hh) * Q + q) * Kl;
  const float* srow = S + rowoff;
  const int*   mrow = mask3d ? (mask + ((long)b * Q + q) * (long)Kl)
                             : (mask + (long)b * Kl);
  __shared__ float red[256];
  const int t = threadIdx.x;
  const int cnt = (Kl + 255) >> 8;
  float lv[4];

  float m = -3.0e38f;
  for (int i = 0; i < cnt; ++i) {
    const int k = t + (i << 8);
    float v = -3.0e38f;
    if (k < Kl) v = (mrow[k] == 0) ? -1e20f : srow[k];
    lv[i] = v;
    m = fmaxf(m, v);
  }
  red[t] = m; __syncthreads();
  for (int s = 128; s > 0; s >>= 1) {
    if (t < s) red[t] = fmaxf(red[t], red[t + s]);
    __syncthreads();
  }
  m = red[0]; __syncthreads();

  float sum = 0.f;
  for (int i = 0; i < cnt; ++i) {
    const int k = t + (i << 8);
    if (k < Kl) { const float e = __expf(lv[i] - m); lv[i] = e; sum += e; }
  }
  red[t] = sum; __syncthreads();
  for (int s = 128; s > 0; s >>= 1) {
    if (t < s) red[t] += red[t + s];
    __syncthreads();
  }
  const float inv = 1.0f / red[0];

  for (int i = 0; i < cnt; ++i) {
    const int k = t + (i << 8);
    if (k < Kl) P[rowoff + k] = (h16)(lv[i] * inv);
  }
}

// ---------------------------------------------------------------------------
// out = LayerNorm(xin + delta) ; also emits f16 copy if x16 != null. D == 1024.
// ---------------------------------------------------------------------------
__global__ __launch_bounds__(256) void add_ln(
    const float* __restrict__ xin, const float* __restrict__ delta,
    const float* __restrict__ g, const float* __restrict__ bb,
    float* __restrict__ xout, h16* __restrict__ x16, int D)
{
  const long row = blockIdx.x;
  const int  t   = threadIdx.x;
  const float* xi = xin   + row * D;
  const float* de = delta + row * D;
  __shared__ float red[256];
  __shared__ float red2[256];
  const int per = D >> 8;   // 4
  float r[4];
  float s = 0.f, s2 = 0.f;
  for (int i = 0; i < per; ++i) {
    const int c = t + (i << 8);
    const float v = xi[c] + de[c];
    r[i] = v; s += v; s2 += v * v;
  }
  red[t] = s; red2[t] = s2; __syncthreads();
  for (int st = 128; st > 0; st >>= 1) {
    if (t < st) { red[t] += red[t + st]; red2[t] += red2[t + st]; }
    __syncthreads();
  }
  const float mean = red[0] / D;
  const float var  = red2[0] / D - mean * mean;
  const float rstd = rsqrtf(var + 1e-5f);
  for (int i = 0; i < per; ++i) {
    const int c = t + (i << 8);
    const float o = g[c] * (r[i] - mean) * rstd + bb[c];
    xout[row * D + c] = o;
    if (x16) x16[row * D + c] = (h16)o;
  }
}

// ---------------------------------------------------------------------------
// Host orchestration
// ---------------------------------------------------------------------------
static void launch_gemm(hipStream_t st,
                        const h16* A, const h16* B, const float* bias,
                        float* C, h16* C16, int M, int N, int K,
                        int lda, int ldbn, int ldbk, int ldc,
                        long sAb, long sAh, long sBb, long sBh,
                        long sCb, long sCh, int Hn, int Z,
                        float alpha, int relu)
{
  dim3 grid(N / BN, M / BM, Z);
  gemm_wmma<<<grid, 256, 0, st>>>(A, B, bias, C, C16, M, N, K,
                                  lda, ldbn, ldbk, ldc,
                                  sAb, sAh, sBb, sBh, sCb, sCh,
                                  Hn, alpha, relu);
}

extern "C" void kernel_launch(void* const* d_in, const int* in_sizes, int n_in,
                              void* d_out, int out_size, void* d_ws, size_t ws_size,
                              hipStream_t stream) {
  (void)in_sizes; (void)n_in; (void)out_size; (void)ws_size;

  const int B = 4, Q = 512, D = 1024, H = 16, F = 4096;
  const int Mx = B * Q;                 // 2048 token rows

  // ---- workspace layout (bump allocator, 256B aligned) ----
  char* ws = (char*)d_ws;
  size_t off = 0;
  auto alloc = [&](size_t bytes) -> char* {
    char* p = ws + off;
    off += (bytes + 255) & ~(size_t)255;
    return p;
  };
  h16*   xf16   = (h16*)  alloc((size_t)Mx * D * 2);            // 4 MB
  float* xres   = (float*)alloc((size_t)Mx * D * 4);            // 8 MB
  float* delta  = (float*)alloc((size_t)Mx * D * 4);            // 8 MB
  h16*   wf16   = (h16*)  alloc((size_t)2 * F * D * 2);         // 16 MB
  h16*   encf16 = (h16*)  alloc((size_t)B * 1024 * D * 2);      // 8 MB
  h16*   qf16   = (h16*)  alloc((size_t)Mx * D * 2);            // 4 MB
  h16*   kf16   = (h16*)  alloc((size_t)B * 1024 * D * 2);      // 8 MB
  h16*   vf16   = (h16*)  alloc((size_t)B * 1024 * D * 2);      // 8 MB
  h16*   vtf16  = (h16*)  alloc((size_t)B * 1024 * D * 2);      // 8 MB
  h16*   of16   = (h16*)  alloc((size_t)Mx * D * 2);            // 4 MB
  float* scores = (float*)alloc((size_t)B * H * Q * 1024 * 4);  // 128 MB
  h16*   probs  = (h16*)  alloc((size_t)B * H * Q * 1024 * 2);  // 64 MB
  h16*   hf16   = (h16*)scores;  // FFN hidden f16 (16 MB) reuses scores region

  // ---- initial state: residual = x (fp32), xf16 = f16(x) ----
  (void)hipMemcpyAsync(xres, d_in[0], (size_t)Mx * D * 4,
                       hipMemcpyDeviceToDevice, stream);
  cvt16<<<2048, 256, 0, stream>>>((const float*)d_in[0], xf16, (long)Mx * D);

  // ---- one MHA block + residual/LN ----
  auto run_mha = [&](const h16* kvsrc, int Mkv, int Kl,
                     const int* mask, int mask3d, int pi,
                     const float* lng, const float* lnb) {
    // convert 4 weight matrices to f16 (wq,wk,wv,wo at pi, pi+2, pi+4, pi+6)
    for (int w = 0; w < 4; ++w)
      cvt16<<<2048, 256, 0, stream>>>((const float*)d_in[pi + 2 * w],
                                      wf16 + (size_t)w * D * D, (long)D * D);
    const h16* wq = wf16;
    const h16* wk = wf16 + (size_t)1 * D * D;
    const h16* wv = wf16 + (size_t)2 * D * D;
    const h16* wo = wf16 + (size_t)3 * D * D;
    const float* bq = (const float*)d_in[pi + 1];
    const float* bk = (const float*)d_in[pi + 3];
    const float* bv = (const float*)d_in[pi + 5];
    const float* bo = (const float*)d_in[pi + 7];

    // q = (x @ wq^T + bq) / sqrt(64)   -> f16 [Mx, D]
    launch_gemm(stream, xf16, wq, bq, nullptr, qf16, Mx, D, D,
                D, D, 1, D, 0, 0, 0, 0, 0, 0, 1, 1, 0.125f, 0);
    // k = kv @ wk^T + bk               -> f16 [Mkv, D]
    launch_gemm(stream, kvsrc, wk, bk, nullptr, kf16, Mkv, D, D,
                D, D, 1, D, 0, 0, 0, 0, 0, 0, 1, 1, 1.0f, 0);
    // v = kv @ wv^T + bv               -> f16 [Mkv, D]
    launch_gemm(stream, kvsrc, wv, bv, nullptr, vf16, Mkv, D, D,
                D, D, 1, D, 0, 0, 0, 0, 0, 0, 1, 1, 1.0f, 0);
    // vt[b,d,k] = v[b,k,d]  (makes the AV GEMM K-contiguous / vectorized)
    transpose_kd<<<dim3(Kl / 32, D / 32, B), 256, 0, stream>>>(vf16, vtf16, Kl, D);

    // scores[b,h,q,k] = q . k  (K = dh = 64), Z = B*H batched
    launch_gemm(stream, qf16, kf16, nullptr, scores, nullptr, Q, Kl, 64,
                D, D, 1, Kl,
                (long)Q * D, 64, (long)Kl * D, 64,
                (long)H * Q * Kl, (long)Q * Kl, H, B * H, 1.0f, 0);

    // masked softmax -> f16 probabilities
    masked_softmax<<<dim3(Q, H, B), 256, 0, stream>>>(scores, mask, probs,
                                                      Kl, Q, mask3d);

    // o[b,q,h*64+d] = probs . v   (N = dh = 64), B from transposed V
    launch_gemm(stream, probs, vtf16, nullptr, nullptr, of16, Q, 64, Kl,
                Kl, Kl, 1, D,
                (long)H * Q * Kl, (long)Q * Kl, (long)D * Kl, (long)64 * Kl,
                (long)Q * D, 64, H, B * H, 1.0f, 0);

    // out = o @ wo^T + bo -> fp32 delta
    launch_gemm(stream, of16, wo, bo, delta, nullptr, Mx, D, D,
                D, D, 1, D, 0, 0, 0, 0, 0, 0, 1, 1, 1.0f, 0);

    // x = LN(x + delta) -> xres (fp32) and xf16
    add_ln<<<Mx, 256, 0, stream>>>(xres, delta, lng, lnb, xres, xf16, D);
  };

  // ---- stage 1: self attention (causal mask, Kl = Q, kv = x) ----
  run_mha(xf16, Mx, Q, (const int*)d_in[1], 1, 8,
          (const float*)d_in[40], (const float*)d_in[41]);

  // ---- stage 2: db cross attention (Kl = 128) ----
  cvt16<<<2048, 256, 0, stream>>>((const float*)d_in[6], encf16, (long)B * 128 * D);
  run_mha(encf16, B * 128, 128, (const int*)d_in[7], 0, 16,
          (const float*)d_in[42], (const float*)d_in[43]);

  // ---- stage 3: kg cross attention (Kl = 128) ----
  cvt16<<<2048, 256, 0, stream>>>((const float*)d_in[4], encf16, (long)B * 128 * D);
  run_mha(encf16, B * 128, 128, (const int*)d_in[5], 0, 24,
          (const float*)d_in[44], (const float*)d_in[45]);

  // ---- stage 4: encoder cross attention (Kl = 1024) ----
  cvt16<<<2048, 256, 0, stream>>>((const float*)d_in[2], encf16, (long)B * 1024 * D);
  run_mha(encf16, B * 1024, 1024, (const int*)d_in[3], 0, 32,
          (const float*)d_in[46], (const float*)d_in[47]);

  // ---- stage 5: FFN ----
  cvt16<<<2048, 256, 0, stream>>>((const float*)d_in[50], wf16, (long)F * D);
  cvt16<<<2048, 256, 0, stream>>>((const float*)d_in[52], wf16 + (size_t)F * D, (long)F * D);
  // h = relu(x @ w1^T + b1) -> f16 [Mx, F]
  launch_gemm(stream, xf16, wf16, (const float*)d_in[51], nullptr, hf16,
              Mx, F, D, D, D, 1, F, 0, 0, 0, 0, 0, 0, 1, 1, 1.0f, 1);
  // delta = h @ w2^T + b2 -> fp32 [Mx, D]
  launch_gemm(stream, hf16, wf16 + (size_t)F * D, (const float*)d_in[53], delta,
              nullptr, Mx, D, F, F, F, 1, D, 0, 0, 0, 0, 0, 0, 1, 1, 1.0f, 0);
  // final: out = LN(x + delta) -> d_out (fp32)
  add_ln<<<Mx, 256, 0, stream>>>(xres, delta, (const float*)d_in[48],
                                 (const float*)d_in[49], (float*)d_out, nullptr, D);
}
